// GRURepresentationModel_74062416053156
// MI455X (gfx1250) — compile-verified, matrix operand order checked by reading, not run on hardware
//
#include <hip/hip_runtime.h>

// MI455X / gfx1250, wave32. Fully fused 2-layer GRU + projection.
// One wave owns a 16-row batch tile and runs the whole T=512 recurrence:
// per step: 24x v_wmma_f32_16x16x32_f16 (x@Wih0, h0@Whh0, h0'@Wih1, h1@Whh1).
// Activations use v_exp_f32 + v_rcp_f32 (TRANS pipe, co-executes with VALU/WMMA).
// x loads are software-pipelined one step ahead to hide VMEM latency.

typedef __attribute__((ext_vector_type(16))) _Float16 v16h;
typedef __attribute__((ext_vector_type(8)))  float    v8f;
typedef __attribute__((ext_vector_type(4)))  float    v4f;

#define T_SEQ 512
#define B_TOT 4096
#define NW    8              // waves per block
#define NTILE 25             // 4 weight matrices * 6 B-tiles + 1 proj tile
#define HPAD  36             // padded row stride (floats) for h scratch in LDS

union U16H { v16h v; _Float16 e[16]; };
union U8F  { v8f  v; float    e[8];  };

// sigmoid/tanh with single-cycle-class rcp instead of IEEE div chains.
__device__ __forceinline__ float sigm(float x) {
  return __builtin_amdgcn_rcpf(1.0f + __expf(-x));
}
__device__ __forceinline__ float tanh_fast(float x) {
  // stable for |x| large: exp->inf => rcp->0 => 1 ; exp->0 => 1-2 = -1
  return 1.0f - 2.0f * __builtin_amdgcn_rcpf(1.0f + __expf(2.0f * x));
}

__device__ __forceinline__ v8f wmma16(v16h a, v16h b, v8f c) {
  // (neg_a, A, neg_b, B, c_mod, C, reuse_a, reuse_b)
  return __builtin_amdgcn_wmma_f32_16x16x32_f16(false, a, false, b, (short)0, c, false, false);
}

// B tiles stored in LDS pre-swizzled to per-lane WMMA order: tile*512 + lane*16 halves.
// (Loop-invariant -> the compiler promotes all 24 recurrence tiles into VGPRs.)
__device__ __forceinline__ v16h load_btile(const _Float16* w, int tile, int lane) {
  U16H r;
  const _Float16* p = w + tile * 512 + lane * 16;
  *(v4f*)&r.e[0] = *(const v4f*)(p);       // 16B
  *(v4f*)&r.e[8] = *(const v4f*)(p + 8);   // 16B
  return r.v;
}

// One GRU timestep for a 16-row tile. xA: input A-matrix (f16).
// hA: recurrent A-matrix (f16, updated). hD: h state in C/D layout (f32, updated).
__device__ __forceinline__ void gru_step(
    const _Float16* ldsW, int ihBase, int hhBase,
    v16h xA, v16h& hA, U8F (&hD)[2],
    const float (&cbi)[6], const float (&cbh)[6],
    float* hrow, int lane)
{
  const int hi = lane >> 4;
  const int m  = lane & 15;
  const int k0 = hi << 3;

  U8F rz[4], xwn[2], ghn[2];
  // r,z gates: xw and gh sum directly -> chain two WMMAs on one accumulator.
  #pragma unroll
  for (int nt = 0; nt < 4; ++nt) {
    float b = cbi[nt] + cbh[nt];
    v8f c = {b, b, b, b, b, b, b, b};
    c = wmma16(xA, load_btile(ldsW, ihBase + nt, lane), c);
    c = wmma16(hA, load_btile(ldsW, hhBase + nt, lane), c);
    rz[nt].v = c;
  }
  // n gate needs xw_n and gh_n separately (gh_n is scaled by r).
  #pragma unroll
  for (int j = 0; j < 2; ++j) {
    float bi = cbi[4 + j], bh = cbh[4 + j];
    v8f ci = {bi, bi, bi, bi, bi, bi, bi, bi};
    v8f ch = {bh, bh, bh, bh, bh, bh, bh, bh};
    xwn[j].v = wmma16(xA, load_btile(ldsW, ihBase + 4 + j, lane), ci);
    ghn[j].v = wmma16(hA, load_btile(ldsW, hhBase + 4 + j, lane), ch);
  }
  // Elementwise gate math in f32, write h to LDS in row-major [16][HPAD].
  #pragma unroll
  for (int jt = 0; jt < 2; ++jt) {
    #pragma unroll
    for (int j = 0; j < 8; ++j) {
      float r = sigm(rz[jt].e[j]);
      float z = sigm(rz[2 + jt].e[j]);
      float n = tanh_fast(__builtin_fmaf(r, ghn[jt].e[j], xwn[jt].e[j]));
      float h = __builtin_fmaf(z, hD[jt].e[j] - n, n);   // (1-z)*n + z*h
      hD[jt].e[j] = h;
      hrow[(j + 8 * hi) * HPAD + jt * 16 + m] = h;       // D-layout scatter
    }
  }
  // Same-wave DS ops execute in order: re-gather as A-matrix (row per lane) + cvt f16.
  U16H nA;
  const float* hp = hrow + m * HPAD + k0;
  v4f a0 = *(const v4f*)(hp);
  v4f a1 = *(const v4f*)(hp + 4);
  v4f a2 = *(const v4f*)(hp + 16);
  v4f a3 = *(const v4f*)(hp + 20);
  #pragma unroll
  for (int i = 0; i < 4; ++i) {
    nA.e[i]      = (_Float16)a0[i];
    nA.e[4 + i]  = (_Float16)a1[i];
    nA.e[8 + i]  = (_Float16)a2[i];
    nA.e[12 + i] = (_Float16)a3[i];
  }
  hA = nA.v;
}

extern "C" __global__ __launch_bounds__(256)
void gru2_fused_kernel(const float* __restrict__ x,
                       const float* __restrict__ Wih0, const float* __restrict__ Whh0,
                       const float* __restrict__ bih0, const float* __restrict__ bhh0,
                       const float* __restrict__ Wih1, const float* __restrict__ Whh1,
                       const float* __restrict__ bih1, const float* __restrict__ bhh1,
                       const float* __restrict__ Wproj, const float* __restrict__ bproj,
                       float* __restrict__ out)
{
  __shared__ __align__(32) _Float16 ldsW[NTILE * 512];   // 25 KB: B-tiles, f16, lane order
  __shared__ __align__(16) float    ldsH[NW][16 * HPAD]; // per-wave h transpose scratch

  const int tid = threadIdx.x;

  // Cooperative one-time weight conversion f32 -> f16 into WMMA B-lane layout.
  // B[k][n] = W[n][k];  lane l -> n = nt*16 + (l&15);  elem e -> k = e + 8*((e>>3)+(l>>4)).
  for (int idx = tid; idx < NTILE * 512; idx += 256) {
    int tile = idx >> 9;
    int rem  = idx & 511;
    int l    = rem >> 4;
    int e    = rem & 15;
    const float* src; int nt;
    if      (tile <  6) { src = Wih0;  nt = tile;      }
    else if (tile < 12) { src = Whh0;  nt = tile - 6;  }
    else if (tile < 18) { src = Wih1;  nt = tile - 12; }
    else if (tile < 24) { src = Whh1;  nt = tile - 18; }
    else                { src = Wproj; nt = 0;         }
    int n = nt * 16 + (l & 15);
    int k = e + 8 * ((e >> 3) + (l >> 4));
    ldsW[idx] = (_Float16)src[n * 32 + k];
  }
  __syncthreads();

  const int lane = tid & 31;
  const int wv   = tid >> 5;
  const int b0   = (blockIdx.x * NW + wv) * 16;   // batch tile base row
  const int m    = lane & 15;
  const int hi   = lane >> 4;
  const int k0   = hi << 3;
  float* hrow = &ldsH[wv][0];

  // Per-lane gate biases (depend only on column n = gate*16 + m).
  float cbi0[6], cbh0[6], cbi1[6], cbh1[6];
  #pragma unroll
  for (int nt = 0; nt < 6; ++nt) {
    cbi0[nt] = bih0[nt * 16 + m];
    cbh0[nt] = bhh0[nt * 16 + m];
    cbi1[nt] = bih1[nt * 16 + m];
    cbh1[nt] = bhh1[nt * 16 + m];
  }

  v16h h0A = {};
  v16h h1A = {};
  v8f  zf  = {};
  U8F h0D[2], h1D[2];
  h0D[0].v = zf; h0D[1].v = zf; h1D[0].v = zf; h1D[1].v = zf;

  const float* xrow = x + (size_t)(b0 + m) * (T_SEQ * 32);

  // Prologue: prefetch x_0 (A-layout rows: two contiguous 8-float runs per lane).
  v4f na0 = *(const v4f*)(xrow + k0);
  v4f na1 = *(const v4f*)(xrow + k0 + 4);
  v4f na2 = *(const v4f*)(xrow + k0 + 16);
  v4f na3 = *(const v4f*)(xrow + k0 + 20);

  for (int t = 0; t < T_SEQ; ++t) {
    // Capture current step's x, then immediately issue next step's loads
    // (clamped index keeps the loop branchless -> EXEC stays all-ones for WMMA).
    v4f a0 = na0, a1 = na1, a2 = na2, a3 = na3;
    int tn = t + 1 < T_SEQ ? t + 1 : t;
    const float* xq = xrow + tn * 32 + k0;
    na0 = *(const v4f*)(xq);
    na1 = *(const v4f*)(xq + 4);
    na2 = *(const v4f*)(xq + 16);
    na3 = *(const v4f*)(xq + 20);

    U16H xA;
    #pragma unroll
    for (int i = 0; i < 4; ++i) {
      xA.e[i]      = (_Float16)a0[i];
      xA.e[4 + i]  = (_Float16)a1[i];
      xA.e[8 + i]  = (_Float16)a2[i];
      xA.e[12 + i] = (_Float16)a3[i];
    }

    gru_step(ldsW,  0,  6, xA.v, h0A, h0D, cbi0, cbh0, hrow, lane);  // layer 0
    gru_step(ldsW, 12, 18, h0A,  h1A, h1D, cbi1, cbh1, hrow, lane);  // layer 1 (fused)
  }

  // Final projection: out[16,16] = h1 @ Wproj^T + b_proj  (one WMMA).
  float bp = bproj[m];
  v8f c = {bp, bp, bp, bp, bp, bp, bp, bp};
  c = wmma16(h1A, load_btile(ldsW, 24, lane), c);
  U8F o; o.v = c;
  #pragma unroll
  for (int j = 0; j < 8; ++j)
    out[(size_t)(b0 + j + 8 * hi) * 16 + m] = o.e[j];
}

extern "C" void kernel_launch(void* const* d_in, const int* in_sizes, int n_in,
                              void* d_out, int out_size, void* d_ws, size_t ws_size,
                              hipStream_t stream) {
  (void)in_sizes; (void)n_in; (void)out_size; (void)d_ws; (void)ws_size;
  const float* x     = (const float*)d_in[0];
  const float* Wih0  = (const float*)d_in[1];
  const float* Whh0  = (const float*)d_in[2];
  const float* bih0  = (const float*)d_in[3];
  const float* bhh0  = (const float*)d_in[4];
  const float* Wih1  = (const float*)d_in[5];
  const float* Whh1  = (const float*)d_in[6];
  const float* bih1  = (const float*)d_in[7];
  const float* bhh1  = (const float*)d_in[8];
  const float* Wproj = (const float*)d_in[9];
  const float* bproj = (const float*)d_in[10];
  float* out = (float*)d_out;

  dim3 grid(B_TOT / (16 * NW));   // 32 blocks x 256 threads = 256 waves = B/16 tiles
  gru2_fused_kernel<<<grid, 256, 0, stream>>>(x, Wih0, Whh0, bih0, bhh0,
                                              Wih1, Whh1, bih1, bhh1,
                                              Wproj, bproj, out);
}
